// Head_54984171323567
// MI455X (gfx1250) — matrix operand order repaired and checked by reading, not compiled, not run on hardware
//
#include <hip/hip_runtime.h>
#include <hip/hip_bf16.h>

typedef __attribute__((ext_vector_type(16))) __bf16 v16bf;
typedef __attribute__((ext_vector_type(8)))  __bf16 v8bf;
typedef __attribute__((ext_vector_type(8)))  float  v8f;

union V16 { v16bf v; v8bf h[2]; };

constexpr int Bv = 8, Tv = 2048, Ev = 512, Hv = 64;

__device__ __forceinline__ v8bf ld8(const __bf16* p) {
    return *reinterpret_cast<const v8bf*>(p);
}

// ---------------------------------------------------------------------------
// Projection: O[row, col] = X[row,:] @ W[:,col] + bias[col], X:[B*T,E] f32,
// W:[E,H] f32. Output bf16, row-major [B*T,H] (STORET=0) or per-batch
// transposed [B,H,T] (STORET=1, used for V).
// ---------------------------------------------------------------------------
template <int STORET>
__global__ __launch_bounds__(256) void proj_wmma(
    const float* __restrict__ X, const float* __restrict__ W,
    const float* __restrict__ bias, __bf16* __restrict__ O)
{
    __shared__ __bf16 Wt[64][520];   // transposed W, padded (row stride 1040B)

    for (int i = threadIdx.x; i < Ev * Hv; i += 256) {
        int e = i >> 6, hcol = i & 63;
        Wt[hcol][e] = (__bf16)W[i];
    }
    __syncthreads();

    const int wave = threadIdx.x >> 5;
    const int lane = threadIdx.x & 31;
    const int lane16 = lane & 15;
    const int hi = lane >> 4;

    const int rowbase = blockIdx.x * 128 + wave * 16;
    const float* xrow = X + (size_t)(rowbase + lane16) * Ev;

    v8f acc[4] = {};   // D layout: VGPR r -> M=r+8*hi, lane16 -> N

    for (int c = 0; c < Ev; c += 32) {
        V16 a;
        float4 f0 = *(const float4*)(xrow + c + 8 * hi);
        float4 f1 = *(const float4*)(xrow + c + 8 * hi + 4);
        float4 f2 = *(const float4*)(xrow + c + 16 + 8 * hi);
        float4 f3 = *(const float4*)(xrow + c + 16 + 8 * hi + 4);
        a.v[0]=(__bf16)f0.x; a.v[1]=(__bf16)f0.y; a.v[2]=(__bf16)f0.z; a.v[3]=(__bf16)f0.w;
        a.v[4]=(__bf16)f1.x; a.v[5]=(__bf16)f1.y; a.v[6]=(__bf16)f1.z; a.v[7]=(__bf16)f1.w;
        a.v[8]=(__bf16)f2.x; a.v[9]=(__bf16)f2.y; a.v[10]=(__bf16)f2.z; a.v[11]=(__bf16)f2.w;
        a.v[12]=(__bf16)f3.x; a.v[13]=(__bf16)f3.y; a.v[14]=(__bf16)f3.z; a.v[15]=(__bf16)f3.w;

        #pragma unroll
        for (int nt = 0; nt < 4; ++nt) {
            V16 bm;
            const __bf16* wp = &Wt[nt * 16 + lane16][c + 8 * hi];
            bm.h[0] = *(const v8bf*)(wp);
            bm.h[1] = *(const v8bf*)(wp + 16);
            acc[nt] = __builtin_amdgcn_wmma_f32_16x16x32_bf16(
                false, a.v, false, bm.v, (short)0, acc[nt], false, false);
        }
    }

    #pragma unroll
    for (int nt = 0; nt < 4; ++nt) {
        int col = nt * 16 + lane16;
        float bvv = bias[col];
        #pragma unroll
        for (int r = 0; r < 8; ++r) {
            int row = rowbase + r + 8 * hi;
            float val = acc[nt][r] + bvv;
            if (STORET == 0) {
                O[(size_t)row * Hv + col] = (__bf16)val;
            } else {
                int bb = row >> 11;          // row / T
                int t  = row & (Tv - 1);     // row % T
                O[((size_t)bb * Hv + col) * Tv + t] = (__bf16)val;
            }
        }
    }
}

// ---------------------------------------------------------------------------
// Flash attention. Block = 8 waves = 128 query rows, walking 32-key chunks
// together. K (32x64 bf16) and V^T (64x32 bf16) chunk tiles are staged into
// double-buffered LDS via async load-to-LDS, issued one chunk ahead so the
// 8 WMMAs + softmax of chunk n hide the latency of chunk n+1's loads.
// Causal gates are forced scalar (readfirstlane) so EXEC is all-ones at WMMA.
// ---------------------------------------------------------------------------
__global__ __launch_bounds__(256) void attn_wmma(
    const __bf16* __restrict__ q, const __bf16* __restrict__ k,
    const __bf16* __restrict__ vt, float* __restrict__ out)
{
    // [0]K0 4KB [1]K1 4KB | V0 4KB V1 4KB | P 8KB  = 24KB
    __shared__ __attribute__((aligned(16))) unsigned char smem[2 * 4096 + 2 * 4096 + 8192];
    __bf16* Klds = (__bf16*)smem;              // two [32][64] buffers
    __bf16* Vlds = (__bf16*)(smem + 8192);     // two [64][32] buffers
    __bf16* Pl   = (__bf16*)(smem + 16384);    // [8][16*32] per-wave P transpose

    const int tid = threadIdx.x;
    const int wave = tid >> 5;
    const int lane = tid & 31;
    const int lane16 = lane & 15;
    const int hi = lane >> 4;

    const int b = blockIdx.y;
    const int qb0 = blockIdx.x * 128;
    const int qbase = qb0 + wave * 16;
    const int qbase_u = __builtin_amdgcn_readfirstlane(qbase);  // SGPR copy for scalar branches

    // q A-operand (A layout: lane16 = row; K runs {16hi..16hi+15})
    V16 qa[2];
    const __bf16* qrow = q + ((size_t)(b * Tv + qbase + lane16)) * Hv;
    #pragma unroll
    for (int cs = 0; cs < 2; ++cs) {
        qa[cs].h[0] = ld8(qrow + cs * 32 + 16 * hi);
        qa[cs].h[1] = ld8(qrow + cs * 32 + 16 * hi + 8);
    }

    float m[8], l[8];
    v8f o[4] = {};
    #pragma unroll
    for (int r = 0; r < 8; ++r) { m[r] = -3.0e38f; l[r] = 0.0f; }

    const __bf16* kbb = k  + (size_t)b * Tv * Hv;
    const __bf16* vbb = vt + (size_t)b * Hv * Tv;
    __bf16* pl = Pl + wave * (16 * 32);

    // per-thread LDS byte addresses (low 32 bits of a generic LDS pointer)
    const uint32_t klds0 = (uint32_t)(size_t)(void*)Klds + (uint32_t)tid * 16u;
    const uint32_t vlds0 = (uint32_t)(size_t)(void*)Vlds +
                           (uint32_t)((tid >> 2) * 64 + (tid & 3) * 16);

    const int nchb = (qb0 + 128) >> 5;   // chunks for the whole block

    auto issue_chunk = [&](int ch) {
        const int jb = ch * 32;
        uint32_t kl = klds0 + (uint32_t)(ch & 1) * 4096u;
        uint32_t vl = vlds0 + (uint32_t)(ch & 1) * 4096u;
        uint32_t kgoff = (uint32_t)((jb * Hv + tid * 8) * 2);                      // bytes
        uint32_t vgoff = (uint32_t)((((tid >> 2) * Tv) + jb + (tid & 3) * 8) * 2); // bytes
        asm volatile("global_load_async_to_lds_b128 %0, %1, %2"
                     :: "v"(kl), "v"(kgoff), "s"(kbb) : "memory");
        asm volatile("global_load_async_to_lds_b128 %0, %1, %2"
                     :: "v"(vl), "v"(vgoff), "s"(vbb) : "memory");
    };

    issue_chunk(0);
    for (int ch = 0; ch < nchb; ++ch) {
        const int jb = ch * 32;
        const __bf16* Kl = Klds + (ch & 1) * 2048;
        const __bf16* Vl = Vlds + (ch & 1) * 2048;

        asm volatile("s_wait_asynccnt 0x0" ::: "memory");
        __syncthreads();                       // chunk ch tiles visible to all waves
        if (ch + 1 < nchb) issue_chunk(ch + 1);  // prefetch into the other buffer

        if (jb < qbase_u + 16) {   // scalar causal gate -> s_cbranch, EXEC stays full
            // ----- scores S = q @ k^T, two 16-key tiles -----
            v8f s0 = {}, s1 = {};
            {
                const __bf16* kp0 = Kl + lane16 * 64 + 16 * hi;
                const __bf16* kp1 = Kl + (16 + lane16) * 64 + 16 * hi;
                V16 kb0a, kb0b, kb1a, kb1b;
                kb0a.h[0] = ld8(kp0);       kb0a.h[1] = ld8(kp0 + 8);
                kb0b.h[0] = ld8(kp0 + 32);  kb0b.h[1] = ld8(kp0 + 40);
                kb1a.h[0] = ld8(kp1);       kb1a.h[1] = ld8(kp1 + 8);
                kb1b.h[0] = ld8(kp1 + 32);  kb1b.h[1] = ld8(kp1 + 40);
                s0 = __builtin_amdgcn_wmma_f32_16x16x32_bf16(false, qa[0].v, false, kb0a.v, (short)0, s0, false, false);
                s0 = __builtin_amdgcn_wmma_f32_16x16x32_bf16(false, qa[1].v, false, kb0b.v, (short)0, s0, false, false);
                s1 = __builtin_amdgcn_wmma_f32_16x16x32_bf16(false, qa[0].v, false, kb1a.v, (short)0, s1, false, false);
                s1 = __builtin_amdgcn_wmma_f32_16x16x32_bf16(false, qa[1].v, false, kb1b.v, (short)0, s1, false, false);
            }
            s0 = s0 * 0.125f;   // 1/sqrt(H)
            s1 = s1 * 0.125f;

            if (jb + 31 > qbase_u) {   // scalar diagonal gate
                #pragma unroll
                for (int r = 0; r < 8; ++r) {
                    int qi = qbase + r + 8 * hi;
                    s0[r] = ((jb + lane16)      > qi) ? -1.0e30f : s0[r];
                    s1[r] = ((jb + 16 + lane16) > qi) ? -1.0e30f : s1[r];
                }
            }

            // ----- online softmax (row r lives in one 16-lane half) -----
            #pragma unroll
            for (int r = 0; r < 8; ++r) {
                float mv = fmaxf(s0[r], s1[r]);
                #pragma unroll
                for (int d = 1; d < 16; d <<= 1) mv = fmaxf(mv, __shfl_xor(mv, d, 16));
                float mn = fmaxf(m[r], mv);
                float al = __expf(m[r] - mn);
                m[r] = mn;
                s0[r] = __expf(s0[r] - mn);
                s1[r] = __expf(s1[r] - mn);
                float rs = s0[r] + s1[r];
                #pragma unroll
                for (int d = 1; d < 16; d <<= 1) rs += __shfl_xor(rs, d, 16);
                l[r] = l[r] * al + rs;
                #pragma unroll
                for (int nt = 0; nt < 4; ++nt) o[nt][r] *= al;
            }

            // ----- transpose P (D layout) -> A layout via per-wave LDS -----
            #pragma unroll
            for (int r = 0; r < 8; ++r) {
                int row = r + 8 * hi;
                pl[row * 32 + lane16]      = (__bf16)s0[r];
                pl[row * 32 + 16 + lane16] = (__bf16)s1[r];
            }
            asm volatile("s_wait_dscnt 0x0" ::: "memory");
            V16 pa;
            pa.h[0] = *(const v8bf*)(pl + lane16 * 32 + 16 * hi);
            pa.h[1] = *(const v8bf*)(pl + lane16 * 32 + 16 * hi + 8);

            // ----- O += P @ V from the shared V^T LDS tile -----
            #pragma unroll
            for (int nt = 0; nt < 4; ++nt) {
                const __bf16* vp = Vl + (nt * 16 + lane16) * 32 + 16 * hi;
                V16 vb;
                vb.h[0] = *(const v8bf*)(vp);
                vb.h[1] = *(const v8bf*)(vp + 8);
                o[nt] = __builtin_amdgcn_wmma_f32_16x16x32_bf16(
                    false, pa.v, false, vb.v, (short)0, o[nt], false, false);
            }
        }
        __syncthreads();   // all reads of buf(ch&1) done before it is refilled
    }

    // ----- normalize and store -----
    #pragma unroll
    for (int nt = 0; nt < 4; ++nt) {
        #pragma unroll
        for (int r = 0; r < 8; ++r) {
            int row = qbase + r + 8 * hi;
            out[((size_t)b * Tv + row) * Hv + nt * 16 + lane16] = o[nt][r] / l[r];
        }
    }
}

// ---------------------------------------------------------------------------
extern "C" void kernel_launch(void* const* d_in, const int* in_sizes, int n_in,
                              void* d_out, int out_size, void* d_ws, size_t ws_size,
                              hipStream_t stream) {
    const float* Q  = (const float*)d_in[0];
    const float* K  = (const float*)d_in[1];
    const float* V  = (const float*)d_in[2];
    const float* Wq = (const float*)d_in[3];
    const float* bq = (const float*)d_in[4];
    const float* Wk = (const float*)d_in[5];
    const float* bk = (const float*)d_in[6];
    const float* Wv = (const float*)d_in[7];
    const float* bv = (const float*)d_in[8];
    // d_in[9] = causal mask, applied analytically in-kernel.

    const size_t nqkv = (size_t)Bv * Tv * Hv;
    __bf16* qb  = (__bf16*)d_ws;                  // bf16 [B,T,H]
    __bf16* kb  = qb + nqkv;                      // bf16 [B,T,H]
    __bf16* vtb = kb + nqkv;                      // bf16 [B,H,T]

    dim3 pg(Bv * Tv / 128);                       // 128 blocks, 8 waves each
    proj_wmma<0><<<pg, 256, 0, stream>>>(Q, Wq, bq, qb);
    proj_wmma<0><<<pg, 256, 0, stream>>>(K, Wk, bk, kb);
    proj_wmma<1><<<pg, 256, 0, stream>>>(V, Wv, bv, vtb);

    dim3 ag(Tv / 128, Bv);                        // 16 x 8 blocks
    attn_wmma<<<ag, 256, 0, stream>>>(qb, kb, vtb, (float*)d_out);
}
